// CausalSelfAttention_87179246174246
// MI455X (gfx1250) — matrix operand order repaired and checked by reading, not compiled, not run on hardware
//
#include <hip/hip_runtime.h>

// Problem dims (fixed by reference)
#define BB 2
#define TT 2048
#define DD 2048
#define HH 16
#define HD 128

typedef __attribute__((ext_vector_type(16))) __bf16 v16bf;
typedef __attribute__((ext_vector_type(8)))  float  v8f;

union FragA {
  v16bf v;
  unsigned short u[16];
  uint4 q[2];
};
union FragC {
  v8f v;
  float f[8];
};

__device__ __forceinline__ unsigned short f2bf(float x) {
  unsigned u = __float_as_uint(x);
  u += 0x7FFFu + ((u >> 16) & 1u);
  return (unsigned short)(u >> 16);
}
__device__ __forceinline__ float bf2f(unsigned short u) {
  return __uint_as_float(((unsigned)u) << 16);
}

// ---------------------------------------------------------------------------
// Kernel 1: f32 -> bf16 conversion of x and W
// ---------------------------------------------------------------------------
__global__ __launch_bounds__(256) void convert_f32_bf16(
    const float* __restrict__ x, const float* __restrict__ W,
    unsigned short* __restrict__ xb, unsigned short* __restrict__ wb,
    int nx, int nw) {
  int i = blockIdx.x * blockDim.x + threadIdx.x;
  int stride = gridDim.x * blockDim.x;
  for (int t = i; t < nx; t += stride) xb[t] = f2bf(x[t]);
  for (int t = i; t < nw; t += stride) wb[t] = f2bf(W[t]);
}

// ---------------------------------------------------------------------------
// Kernel 2/5: bf16 GEMM  C[M,N] = A[M,K] * Bm[N,K]^T   (Bm row-major over K)
// Wave tile: 32(M) x 64(N) -> 8 accumulators, B fragments reused 2x.
// Block: 8 waves (2 M x 4 N) -> 64(M) x 256(N).
// ---------------------------------------------------------------------------
template <bool OUT_F32>
__global__ __launch_bounds__(256) void gemm_bf16(
    const unsigned short* __restrict__ A, const unsigned short* __restrict__ Bm,
    void* __restrict__ C, int M, int N, int K) {
  const int wave = threadIdx.x >> 5;
  const int lane = threadIdx.x & 31;
  const int hf = lane >> 4;   // lane half (0/1)
  const int l15 = lane & 15;
  const int mwave = wave >> 2;  // 0..1
  const int nwave = wave & 3;   // 0..3
  const int m0 = blockIdx.x * 64 + mwave * 32;
  const int n0 = blockIdx.y * 256 + nwave * 64;

  FragC acc[2][4];
#pragma unroll
  for (int i = 0; i < 2; ++i)
#pragma unroll
    for (int j = 0; j < 4; ++j)
#pragma unroll
      for (int r = 0; r < 8; ++r) acc[i][j].f[r] = 0.f;

  const unsigned short* Arow0 = A + (size_t)(m0 + l15) * K;
  const unsigned short* Arow1 = A + (size_t)(m0 + 16 + l15) * K;
  const int kbase = hf * 8;

  for (int k = 0; k < K; k += 32) {
    FragA a0, a1;
    a0.q[0] = *(const uint4*)(Arow0 + k + kbase);       // K = kbase..kbase+7
    a0.q[1] = *(const uint4*)(Arow0 + k + kbase + 16);  // K = kbase+16..+23
    a1.q[0] = *(const uint4*)(Arow1 + k + kbase);
    a1.q[1] = *(const uint4*)(Arow1 + k + kbase + 16);
#pragma unroll
    for (int j = 0; j < 4; ++j) {
      FragA b;
      const unsigned short* Brow =
          Bm + (size_t)(n0 + j * 16 + l15) * K + k + hf * 8;
      b.q[0] = *(const uint4*)(Brow);
      b.q[1] = *(const uint4*)(Brow + 16);
      acc[0][j].v = __builtin_amdgcn_wmma_f32_16x16x32_bf16(
          false, a0.v, false, b.v, (short)0, acc[0][j].v, false, false);
      acc[1][j].v = __builtin_amdgcn_wmma_f32_16x16x32_bf16(
          false, a1.v, false, b.v, (short)0, acc[1][j].v, false, false);
    }
  }

#pragma unroll
  for (int i = 0; i < 2; ++i)
#pragma unroll
    for (int j = 0; j < 4; ++j)
#pragma unroll
      for (int r = 0; r < 8; ++r) {
        const int m = m0 + i * 16 + hf * 8 + r;
        const int n = n0 + j * 16 + l15;
        if (OUT_F32)
          ((float*)C)[(size_t)m * N + n] = acc[i][j].f[r];
        else
          ((unsigned short*)C)[(size_t)m * N + n] = f2bf(acc[i][j].f[r]);
      }
}

// ---------------------------------------------------------------------------
// Kernel 3: per-head RMS norm + rotary; scatter Q,K -> (B,H,T,HD) bf16,
//           V -> (B,H,HD,T) bf16 (transposed for PV fragment loads).
// One wave handles one (b, t, head-group) of 128 values.
// ---------------------------------------------------------------------------
__global__ __launch_bounds__(256) void normrope(
    const unsigned short* __restrict__ qkv, const float* __restrict__ cosb,
    const float* __restrict__ sinb, unsigned short* __restrict__ qh,
    unsigned short* __restrict__ kh, unsigned short* __restrict__ vh) {
  const int wave = threadIdx.x >> 5;
  const int lane = threadIdx.x & 31;
  const int idx = blockIdx.x * 8 + wave;  // 0 .. B*T*48-1
  const int row = idx / 48;               // b*T + t
  const int hg = idx - row * 48;          // 0..47  (q:0-15 k:16-31 v:32-47)
  const int b = row >> 11;
  const int t = row & (TT - 1);

  const unsigned short* src = qkv + (size_t)row * (3 * HH * HD) + hg * HD;
  float e0 = bf2f(src[lane]);
  float e1 = bf2f(src[lane + 32]);
  float e2 = bf2f(src[lane + 64]);
  float e3 = bf2f(src[lane + 96]);

  if (hg < 32) {
    // RMS norm over HD=128
    float ss = e0 * e0 + e1 * e1 + e2 * e2 + e3 * e3;
#pragma unroll
    for (int m = 16; m >= 1; m >>= 1) ss += __shfl_xor(ss, m, 32);
    const float rinv = rsqrtf(ss * (1.f / 128.f) + 1.1920929e-7f);
    e0 *= rinv; e1 *= rinv; e2 *= rinv; e3 *= rinv;
    // Rotary: pairs (i, i+64) with cos/sin[t, i], i = lane and lane+32
    const float c0 = cosb[t * 64 + lane],      s0 = sinb[t * 64 + lane];
    const float c1 = cosb[t * 64 + lane + 32], s1 = sinb[t * 64 + lane + 32];
    const float y0 =  e0 * c0 + e2 * s0;
    const float y2 = -e0 * s0 + e2 * c0;
    const float y1 =  e1 * c1 + e3 * s1;
    const float y3 = -e1 * s1 + e3 * c1;
    const int h = hg & 15;
    unsigned short* dst =
        (hg < 16 ? qh : kh) + ((size_t)(b * HH + h) * TT + t) * HD;
    dst[lane]      = f2bf(y0);
    dst[lane + 32] = f2bf(y1);
    dst[lane + 64] = f2bf(y2);
    dst[lane + 96] = f2bf(y3);
  } else {
    const int h = hg - 32;
    unsigned short* dst = vh + (size_t)(b * HH + h) * HD * TT + t;
    dst[(size_t)(lane)      * TT] = src[lane];
    dst[(size_t)(lane + 32) * TT] = src[lane + 32];
    dst[(size_t)(lane + 64) * TT] = src[lane + 64];
    dst[(size_t)(lane + 96) * TT] = src[lane + 96];
  }
}

// ---------------------------------------------------------------------------
// Kernel 4: flash attention. Block = (b, h, 128 q rows); 8 waves x 16 q rows.
// ---------------------------------------------------------------------------
__global__ __launch_bounds__(256) void attn(
    const unsigned short* __restrict__ qh, const unsigned short* __restrict__ kh,
    const unsigned short* __restrict__ vh, const float* __restrict__ scale_p,
    unsigned short* __restrict__ yb) {
  __shared__ unsigned short Kl[32 * 136];      // 32 keys x 128 feats (pad 136)
  __shared__ unsigned short Vl[128 * 40];      // 128 hd   x 32 keys  (pad 40)
  __shared__ unsigned short Pl[8 * 16 * 40];   // per-wave 16 x 32 P (pad 40)

  const float scale = scale_p[0];
  const int tid = threadIdx.x;
  const int wave = tid >> 5, lane = tid & 31;
  const int hf = lane >> 4, l15 = lane & 15;
  const int qb = blockIdx.x & 15;
  const int bh = blockIdx.x >> 4;  // b*H + h
  const int b = bh >> 4, h = bh & 15;
  const int q0blk = qb * 128;
  const int q0w = q0blk + wave * 16;

  const unsigned short* Qg = qh + (size_t)bh * TT * HD;
  const unsigned short* Kg = kh + (size_t)bh * TT * HD;
  const unsigned short* Vg = vh + (size_t)bh * HD * TT;

  // Q A-fragments for the wave's 16 rows, K-dim = 128 -> 4 chunks of 32
  FragA qa[4];
  {
    const unsigned short* Qrow = Qg + (size_t)(q0w + l15) * HD + hf * 8;
#pragma unroll
    for (int c = 0; c < 4; ++c) {
      qa[c].q[0] = *(const uint4*)(Qrow + c * 32);
      qa[c].q[1] = *(const uint4*)(Qrow + c * 32 + 16);
    }
  }

  FragC acc[8];
#pragma unroll
  for (int c = 0; c < 8; ++c)
#pragma unroll
    for (int r = 0; r < 8; ++r) acc[c].f[r] = 0.f;
  float mrow[8], lrow[8];
#pragma unroll
  for (int r = 0; r < 8; ++r) { mrow[r] = -3.0e38f; lrow[r] = 0.f; }

  unsigned short* Pw = Pl + wave * 16 * 40;
  const int kend = q0blk + 128;  // causal upper bound for this block

  // per-thread staging coordinates
  const int rowk = tid >> 3, segk = tid & 7;  // K tile: 32 rows x 8 segs
  const int rowv = tid >> 1, segv = tid & 1;  // V tile: 128 rows x 2 segs

  for (int kb = 0; kb < kend; kb += 32) {
    __syncthreads();
    {  // stage K block: 32 rows x 128 feats
      const uint4* s = (const uint4*)(Kg + (size_t)(kb + rowk) * HD + segk * 16);
      uint4* d = (uint4*)(Kl + rowk * 136 + segk * 16);
      d[0] = s[0]; d[1] = s[1];
    }
    {  // stage V block (transposed source): 128 hd-rows x 32 keys
      const uint4* s = (const uint4*)(Vg + (size_t)rowv * TT + kb + segv * 16);
      uint4* d = (uint4*)(Vl + rowv * 40 + segv * 16);
      d[0] = s[0]; d[1] = s[1];
    }
    __syncthreads();

    // Prefetch next key block's K/V lines while matrix pipes work
    if (kb + 32 < kend) {
      __builtin_prefetch(Kg + (size_t)(kb + 32 + rowk) * HD + segk * 16, 0, 3);
      __builtin_prefetch(Vg + (size_t)rowv * TT + kb + 32 + segv * 16, 0, 3);
    }

    // S = Q * K^T for two 16-key subtiles
    FragC s0, s1;
#pragma unroll
    for (int r = 0; r < 8; ++r) { s0.f[r] = 0.f; s1.f[r] = 0.f; }
#pragma unroll
    for (int c = 0; c < 4; ++c) {
      FragA bk0, bk1;
      const unsigned short* kp0 = Kl + l15 * 136 + c * 32 + hf * 8;
      const unsigned short* kp1 = Kl + (16 + l15) * 136 + c * 32 + hf * 8;
      bk0.q[0] = *(const uint4*)(kp0); bk0.q[1] = *(const uint4*)(kp0 + 16);
      bk1.q[0] = *(const uint4*)(kp1); bk1.q[1] = *(const uint4*)(kp1 + 16);
      s0.v = __builtin_amdgcn_wmma_f32_16x16x32_bf16(
          false, qa[c].v, false, bk0.v, (short)0, s0.v, false, false);
      s1.v = __builtin_amdgcn_wmma_f32_16x16x32_bf16(
          false, qa[c].v, false, bk1.v, (short)0, s1.v, false, false);
    }

    // online softmax (C layout: lane = key column, element r -> row hf*8+r)
    float p0[8], p1[8];
#pragma unroll
    for (int r = 0; r < 8; ++r) {
      const int qrow = q0w + hf * 8 + r;
      float a0 = s0.f[r] * scale;
      float a1 = s1.f[r] * scale;
      if (kb + l15 > qrow)      a0 = -3.0e38f;
      if (kb + 16 + l15 > qrow) a1 = -3.0e38f;
      float mx = fmaxf(a0, a1);
      mx = fmaxf(mx, __shfl_xor(mx, 1, 32));
      mx = fmaxf(mx, __shfl_xor(mx, 2, 32));
      mx = fmaxf(mx, __shfl_xor(mx, 4, 32));
      mx = fmaxf(mx, __shfl_xor(mx, 8, 32));
      const float mnew = fmaxf(mrow[r], mx);
      const float corr = __expf(mrow[r] - mnew);
      mrow[r] = mnew;
      p0[r] = __expf(a0 - mnew);
      p1[r] = __expf(a1 - mnew);
      float rs = p0[r] + p1[r];
      rs += __shfl_xor(rs, 1, 32);
      rs += __shfl_xor(rs, 2, 32);
      rs += __shfl_xor(rs, 4, 32);
      rs += __shfl_xor(rs, 8, 32);
      lrow[r] = lrow[r] * corr + rs;
#pragma unroll
      for (int c = 0; c < 8; ++c) acc[c].f[r] *= corr;
    }

    // C-layout -> A-layout transpose of P through per-wave LDS (as bf16)
#pragma unroll
    for (int r = 0; r < 8; ++r) {
      Pw[(hf * 8 + r) * 40 + l15]      = f2bf(p0[r]);
      Pw[(hf * 8 + r) * 40 + 16 + l15] = f2bf(p1[r]);
    }
    asm volatile("s_wait_dscnt 0" ::: "memory");

    FragA pa;
    {
      const unsigned short* pp = Pw + l15 * 40 + hf * 8;
      pa.q[0] = *(const uint4*)(pp);
      pa.q[1] = *(const uint4*)(pp + 16);
    }

    // Y += P * V   (8 hd chunks of 16)
#pragma unroll
    for (int c = 0; c < 8; ++c) {
      FragA vb;
      const unsigned short* vp = Vl + (c * 16 + l15) * 40 + hf * 8;
      vb.q[0] = *(const uint4*)(vp);
      vb.q[1] = *(const uint4*)(vp + 16);
      acc[c].v = __builtin_amdgcn_wmma_f32_16x16x32_bf16(
          false, pa.v, false, vb.v, (short)0, acc[c].v, false, false);
    }
  }

  // normalize and write y as bf16 in (B, T, H*HD)
#pragma unroll
  for (int c = 0; c < 8; ++c)
#pragma unroll
    for (int r = 0; r < 8; ++r) {
      const int qrow = q0w + hf * 8 + r;
      const float v = acc[c].f[r] / lrow[r];
      yb[((size_t)b * TT + qrow) * DD + h * HD + c * 16 + l15] = f2bf(v);
    }
}

// ---------------------------------------------------------------------------
// Host launcher
// ---------------------------------------------------------------------------
extern "C" void kernel_launch(void* const* d_in, const int* in_sizes, int n_in,
                              void* d_out, int out_size, void* d_ws,
                              size_t ws_size, hipStream_t stream) {
  const float* x = (const float*)d_in[0];     // (B,T,D)
  const float* W = (const float*)d_in[1];     // (4,D,D)
  const float* cosb = (const float*)d_in[2];  // (T,64)
  const float* sinb = (const float*)d_in[3];  // (T,64)
  const float* scale = (const float*)d_in[4]; // scalar
  float* out = (float*)d_out;                 // (B,T,D) f32

  const size_t NX = (size_t)BB * TT * DD;        // 8,388,608
  const size_t NW = (size_t)4 * DD * DD;         // 16,777,216
  const size_t NQKV = (size_t)BB * TT * 3 * DD;  // 25,165,824
  const size_t NHEAD = (size_t)BB * HH * TT * HD;// 8,388,608

  unsigned short* ws = (unsigned short*)d_ws;
  unsigned short* xb = ws;
  unsigned short* wb = xb + NX;
  unsigned short* qkvb = wb + NW;
  unsigned short* qhh = qkvb + NQKV;
  unsigned short* khh = qhh + NHEAD;
  unsigned short* vhh = khh + NHEAD;
  unsigned short* ybb = vhh + NHEAD;

  convert_f32_bf16<<<8192, 256, 0, stream>>>(x, W, xb, wb, (int)NX, (int)NW);

  // QKV projection: M=4096, N=6144, K=2048
  dim3 g2(4096 / 64, 6144 / 256);
  gemm_bf16<false><<<g2, 256, 0, stream>>>(xb, wb, qkvb, 4096, 6144, 2048);

  // RMS norm + rotary + layout
  normrope<<<(BB * TT * 48) / 8, 256, 0, stream>>>(qkvb, cosb, sinb, qhh, khh,
                                                   vhh);

  // Flash attention
  attn<<<BB * HH * (TT / 128), 256, 0, stream>>>(qhh, khh, vhh, scale, ybb);

  // Output projection with W[3]: M=4096, N=2048, K=2048
  dim3 g5(4096 / 64, 2048 / 256);
  gemm_bf16<true><<<g5, 256, 0, stream>>>(ybb, wb + (size_t)3 * DD * DD, out,
                                          4096, 2048, 2048);
}
